// DIN_18476949307885
// MI455X (gfx1250) — compile-verified, matrix operand order checked by reading, not run on hardware
//
#include <hip/hip_runtime.h>
#include <hip/hip_bf16.h>
#include <math.h>

// ---------------------------------------------------------------------------
// DIN forward for MI455X (gfx1250, wave32, WMMA bf16, f32 accumulate).
//
// Input flatten order (setup_inputs insertion order):
//  0 user_emb[100000*32] 1 item_emb[100000*32]
//  2..6   att_mlp[0]: W[128*64] b[64] alpha[64] gamma[64] beta[64]
//  7..11  att_mlp[1]: W[64*32]  b[32] alpha[32] gamma[32] beta[32]
// 12..16  att_mlp[2]: W[32*16]  b[16] alpha[16] gamma[16] beta[16]
// 17,18   att_out: W[16] b[1]
// 19..23  fc_mlp[0]: W[96*200] b[200] alpha[200] gamma[200] beta[200]
// 24..28  fc_mlp[1]: W[200*80] b[80]  alpha[80]  gamma[80]  beta[80]
// 29,30   fc_out: W[80] b[1]
// 31 user_ids[8192] 32 hist_item_ids[8192*200] 33 hist_lens[8192] 34 target[8192]
// ---------------------------------------------------------------------------

#define BB   8192
#define LL   200
#define DD   32
#define BL   (BB * LL)        // 1,638,400 rows, multiple of 16
#define EPSF 1e-8f

typedef __attribute__((ext_vector_type(8)))  float  v8f;
typedef __attribute__((ext_vector_type(16))) __bf16 v16bf;

union Frag { v16bf v; uint4 q[2]; };     // one WMMA 16-bit operand (32B/lane)

static __device__ __forceinline__ unsigned short f2bf(float f) {
  union { float f; unsigned u; } c; c.f = f;
  unsigned u = c.u;
  unsigned r = (u + 0x7FFFu + ((u >> 16) & 1u)) >> 16;   // RNE
  return (unsigned short)r;
}
static __device__ __forceinline__ unsigned pack2(float a, float b) {
  return (unsigned)f2bf(a) | ((unsigned)f2bf(b) << 16);
}

// Forward map (CDNA5 ISA, 16-bit A/B layout):
// half-element h of a lane holds K = base(h>>1) + 8*laneHi + (h&1),
// base(v) = 2v (v<4) else 2v+8.
static __device__ __forceinline__ int wmma_k(int h, int laneHi) {
  int v = h >> 1, lo = h & 1;
  int base = (v < 4) ? (v * 2) : (v * 2 + 8);
  return base + (laneHi ? 8 : 0) + lo;
}
// Inverse map for EVEN koff within a 32-wide k-chunk: returns the ushort index
// (fragLane*16 + h) inside a fragment where (row, koff) lives; koff+1 -> +1.
static __device__ __forceinline__ int frag_pos_even(int koff, int row) {
  int hi  = (koff >> 3) & 1;
  int grp = koff >> 4;
  int wi  = koff & 7;
  int v   = (grp ? 4 : 0) + (wi >> 1);
  return (row + hi * 16) * 16 + v * 2;
}

static __device__ __forceinline__ float dice_apply(float x, float2 st,
                                                   float g, float be, float al) {
  float xn = (x - st.x) * st.y * g + be;
  float s  = 1.0f / (1.0f + __expf(-xn));
  return x * (s + (1.0f - s) * al);
}

// ---------------------------------------------------------------------------
// Weight prep: write every weight matrix directly in WMMA fragment order:
// element((kc,nt), lane, h) at [( (kc*NT+nt)*32 + lane )*16 + h], so a B
// fragment is 2x global_load_b128 per lane with zero packing moves.
// Layer-0 concat fold (A = [e_h | e_h*e_a | e_a], K=96):
//   rows  0..31 = W0a+W0c   (e_h path)
//   rows 32..63 = W0d       (e_h*e_a path)
//   rows 64..95 = W0b-W0c   (e_a path; replaces the per-batch bias gather)
// ---------------------------------------------------------------------------
__global__ void din_prepw_kernel(const float* __restrict__ W0,
                                 const float* __restrict__ W1,
                                 const float* __restrict__ W2,
                                 const float* __restrict__ Wf0,
                                 const float* __restrict__ Wf1,
                                 unsigned short* __restrict__ Wcomb,  // 3x4 frags
                                 unsigned short* __restrict__ W1b,    // 2x2
                                 unsigned short* __restrict__ W2b,    // 1x1
                                 unsigned short* __restrict__ Wf0b,   // 3x13
                                 unsigned short* __restrict__ Wf1b) { // 7x5
  int t = threadIdx.x;
  for (int i = t; i < 3 * 4 * 512; i += 256) {
    int frag = i >> 9, w = i & 511, lane = w >> 4, h = w & 15;
    int kc = frag >> 2, nt = frag & 3;
    int k = kc * 32 + wmma_k(h, lane >> 4), n = nt * 16 + (lane & 15);
    float v;
    if (k < 32)       v = W0[k * 64 + n] + W0[(64 + k) * 64 + n];
    else if (k < 64)  v = W0[(96 + (k - 32)) * 64 + n];
    else              v = W0[(32 + (k - 64)) * 64 + n] - W0[(64 + (k - 64)) * 64 + n];
    Wcomb[i] = f2bf(v);
  }
  for (int i = t; i < 2 * 2 * 512; i += 256) {
    int frag = i >> 9, w = i & 511, lane = w >> 4, h = w & 15;
    int kc = frag >> 1, nt = frag & 1;
    int k = kc * 32 + wmma_k(h, lane >> 4), n = nt * 16 + (lane & 15);
    W1b[i] = f2bf(W1[k * 32 + n]);
  }
  for (int i = t; i < 512; i += 256) {
    int lane = i >> 4, h = i & 15;
    int k = wmma_k(h, lane >> 4), n = lane & 15;
    W2b[i] = f2bf(W2[k * 16 + n]);
  }
  for (int i = t; i < 3 * 13 * 512; i += 256) {
    int frag = i >> 9, w = i & 511, lane = w >> 4, h = w & 15;
    int kc = frag / 13, nt = frag % 13;
    int k = kc * 32 + wmma_k(h, lane >> 4), n = nt * 16 + (lane & 15);
    Wf0b[i] = f2bf(n < 200 ? Wf0[k * 200 + n] : 0.0f);   // pad N 200 -> 208
  }
  for (int i = t; i < 7 * 5 * 512; i += 256) {
    int frag = i >> 9, w = i & 511, lane = w >> 4, h = w & 15;
    int kc = frag / 5, nt = frag % 5;
    int k = kc * 32 + wmma_k(h, lane >> 4), n = nt * 16 + (lane & 15);
    Wf1b[i] = f2bf(k < 200 ? Wf1[k * 80 + n] : 0.0f);    // pad K 200 -> 224
  }
}

// ---------------------------------------------------------------------------
// Attention layer 0: gather e_h/e_a (float4 loads), build A=[e_h|e_h*e_a|e_a]
// (K=96) in fragment-ordered LDS, 3 k-steps x 4 N-tiles of
// v_wmma_f32_16x16x32_bf16, + uniform bias b0. One wave / 16-row tile.
// ---------------------------------------------------------------------------
__global__ void din_att0_kernel(const float* __restrict__ item_emb,
                                const int* __restrict__ hist,
                                const int* __restrict__ tgt,
                                const unsigned short* __restrict__ Wcomb,
                                const float* __restrict__ b0,      // [64]
                                float* __restrict__ Y0) {          // [BL][64]
  __shared__ unsigned short ldsA[8 * 1536];   // per wave: 3 chunks x 512
  const int wave = threadIdx.x >> 5, lane = threadIdx.x & 31;
  const int laneLo = lane & 15, laneHi = lane >> 4;
  const int tile = blockIdx.x * 8 + wave;
  const int r0 = tile * 16;
  const int bt0 = r0 / LL;                 // batch of first row in tile
  const int bbound = (bt0 + 1) * LL;       // first row of next batch
  unsigned short* myA = ldsA + wave * 1536;

  {   // stage: lane -> (row, half-of-32); vector gathers, packed LDS stores
    int row = lane >> 1, half = lane & 1;
    int r = r0 + row;
    int bidx = (r >= bbound) ? (bt0 + 1) : bt0;   // only division-free compare
    const float4* eh4 = (const float4*)(item_emb + (size_t)hist[r] * 32 + half * 16);
    const float4* ea4 = (const float4*)(item_emb + (size_t)tgt[bidx] * 32 + half * 16);
    union { float4 q[4]; float f[16]; } eh, ea;
    eh.q[0] = eh4[0]; eh.q[1] = eh4[1]; eh.q[2] = eh4[2]; eh.q[3] = eh4[3];
    ea.q[0] = ea4[0]; ea.q[1] = ea4[1]; ea.q[2] = ea4[2]; ea.q[3] = ea4[3];
    #pragma unroll
    for (int jp = 0; jp < 8; ++jp) {
      int j = jp * 2;
      int pos = frag_pos_even(half * 16 + j, row);
      *(unsigned*)(myA + pos)        = pack2(eh.f[j], eh.f[j + 1]);
      *(unsigned*)(myA + 512 + pos)  = pack2(eh.f[j] * ea.f[j],
                                             eh.f[j + 1] * ea.f[j + 1]);
      *(unsigned*)(myA + 1024 + pos) = pack2(ea.f[j], ea.f[j + 1]);
    }
  }
  __syncthreads();

  Frag a0, a1, a2;
  {   // 2x ds_load_b128 per fragment
    const uint4* ap0 = (const uint4*)myA;
    const uint4* ap1 = (const uint4*)(myA + 512);
    const uint4* ap2 = (const uint4*)(myA + 1024);
    a0.q[0] = ap0[lane * 2]; a0.q[1] = ap0[lane * 2 + 1];
    a1.q[0] = ap1[lane * 2]; a1.q[1] = ap1[lane * 2 + 1];
    a2.q[0] = ap2[lane * 2]; a2.q[1] = ap2[lane * 2 + 1];
  }

  const uint4* bw = (const uint4*)Wcomb;
  #pragma unroll
  for (int nt = 0; nt < 4; ++nt) {
    Frag b0f, b1f, b2f;
    b0f.q[0] = bw[((0 + nt) * 32 + lane) * 2];
    b0f.q[1] = bw[((0 + nt) * 32 + lane) * 2 + 1];
    b1f.q[0] = bw[((4 + nt) * 32 + lane) * 2];
    b1f.q[1] = bw[((4 + nt) * 32 + lane) * 2 + 1];
    b2f.q[0] = bw[((8 + nt) * 32 + lane) * 2];
    b2f.q[1] = bw[((8 + nt) * 32 + lane) * 2 + 1];
    v8f c = {0.f, 0.f, 0.f, 0.f, 0.f, 0.f, 0.f, 0.f};
    c = __builtin_amdgcn_wmma_f32_16x16x32_bf16(false, a0.v, false, b0f.v,
                                                (short)0, c, false, false);
    c = __builtin_amdgcn_wmma_f32_16x16x32_bf16(false, a1.v, false, b1f.v,
                                                (short)0, c, false, false);
    c = __builtin_amdgcn_wmma_f32_16x16x32_bf16(false, a2.v, false, b2f.v,
                                                (short)0, c, false, false);
    int col = nt * 16 + laneLo;
    float bv = b0[col];                      // uniform bias, 1 load per n-tile
    #pragma unroll
    for (int i = 0; i < 8; ++i) {
      int rr = r0 + i + laneHi * 8;
      Y0[(size_t)rr * 64 + col] = c[i] + bv;
    }
  }
}

// ---------------------------------------------------------------------------
// Generic fused [optional dice] -> GEMM (bf16 WMMA, f32 acc) -> +bias.
// KPAD is compile-time: constant-divide staging, fully unrolled k-chunk loop.
// A: 16 x KPAD staged per-wave in fragment-ordered dynamic LDS.
// B: pre-swizzled fragments in ws (2x global_load_b128 per fragment).
// ---------------------------------------------------------------------------
template <int NTILES, int KPAD>
__global__ void din_dice_gemm_kernel(const float* __restrict__ X, int Kreal,
                                     const float2* __restrict__ stats,
                                     const float* __restrict__ gamma,
                                     const float* __restrict__ beta,
                                     const float* __restrict__ alpha,
                                     const unsigned short* __restrict__ Wbf,
                                     const float* __restrict__ bias,
                                     float* __restrict__ Y, int Nreal) {
  extern __shared__ unsigned short smem[];
  const int wave = threadIdx.x >> 5, lane = threadIdx.x & 31;
  const int laneLo = lane & 15, laneHi = lane >> 4;
  unsigned short* ldsA = smem + wave * 16 * KPAD;   // kchunks x 512 ushorts
  const int tile = blockIdx.x * 8 + wave;
  const int r0 = tile * 16;

  // stage A (fragment order): consecutive lanes read consecutive float pairs
  constexpr int Kh = KPAD >> 1;
  for (int idx = lane; idx < 16 * Kh; idx += 32) {
    int row = idx / Kh;                 // constant divide -> mul/shift
    int k = (idx - row * Kh) * 2;
    float x0 = 0.0f, x1 = 0.0f;
    if (k < Kreal) {
      x0 = X[(size_t)(r0 + row) * Kreal + k];
      if (stats) x0 = dice_apply(x0, stats[k], gamma[k], beta[k], alpha[k]);
    }
    if (k + 1 < Kreal) {
      x1 = X[(size_t)(r0 + row) * Kreal + k + 1];
      if (stats) x1 = dice_apply(x1, stats[k + 1], gamma[k + 1], beta[k + 1], alpha[k + 1]);
    }
    int kc = k >> 5;
    int pos = frag_pos_even(k & 31, row);
    *(unsigned*)(ldsA + kc * 512 + pos) = pack2(x0, x1);
  }
  __syncthreads();

  v8f acc[NTILES];
  #pragma unroll
  for (int nt = 0; nt < NTILES; ++nt)
    acc[nt] = (v8f){0.f, 0.f, 0.f, 0.f, 0.f, 0.f, 0.f, 0.f};

  const uint4* bw = (const uint4*)Wbf;
  constexpr int kchunks = KPAD >> 5;
  #pragma unroll
  for (int kc = 0; kc < kchunks; ++kc) {
    if (kc + 1 < kchunks)                       // global_prefetch_b8 next B block
      __builtin_prefetch(Wbf + (size_t)(kc + 1) * NTILES * 512, 0, 1);
    Frag a;
    const uint4* ap = (const uint4*)(ldsA + kc * 512);
    a.q[0] = ap[lane * 2];
    a.q[1] = ap[lane * 2 + 1];
    #pragma unroll
    for (int nt = 0; nt < NTILES; ++nt) {
      Frag bf;
      size_t fi = ((size_t)(kc * NTILES + nt) * 32 + lane) * 2;
      bf.q[0] = bw[fi];
      bf.q[1] = bw[fi + 1];
      acc[nt] = __builtin_amdgcn_wmma_f32_16x16x32_bf16(false, a.v, false, bf.v,
                                                        (short)0, acc[nt],
                                                        false, false);
    }
  }

  #pragma unroll
  for (int nt = 0; nt < NTILES; ++nt) {
    int col = nt * 16 + laneLo;
    if (col < Nreal) {
      float bv = bias[col];
      #pragma unroll
      for (int i = 0; i < 8; ++i) {
        int rr = r0 + i + laneHi * 8;
        Y[(size_t)rr * Nreal + col] = acc[nt][i] + bv;
      }
    }
  }
}

// ---------------------------------------------------------------------------
// Deterministic per-column mean/rstd: stage 1 chunk partials, stage 2 finish.
// ---------------------------------------------------------------------------
__global__ void din_colstats_part_kernel(const float* __restrict__ Y,
                                         int rows, int cols, int CH,
                                         float2* __restrict__ partial) {
  int chunk = blockIdx.x, c = threadIdx.x;
  if (c >= cols) return;
  int rpc = rows / CH, rbeg = chunk * rpc;
  float s = 0.f, q = 0.f;
  for (int r = rbeg; r < rbeg + rpc; ++r) {
    float v = Y[(size_t)r * cols + c];
    s += v; q += v * v;
  }
  partial[(size_t)chunk * cols + c] = make_float2(s, q);
}

__global__ void din_colstats_final_kernel(const float2* __restrict__ partial,
                                          int CH, int cols, float invRows,
                                          float2* __restrict__ stats) {
  int c = threadIdx.x;
  if (c >= cols) return;
  float s = 0.f, q = 0.f;
  for (int ch = 0; ch < CH; ++ch) {
    float2 p = partial[(size_t)ch * cols + c];
    s += p.x; q += p.y;
  }
  float mean = s * invRows;
  float var  = q * invRows - mean * mean;
  stats[c] = make_float2(mean, rsqrtf(var + EPSF));
}

// ---------------------------------------------------------------------------
// Attention scores: dice on Y2 (vector loads), dot att_out, mask.
// ---------------------------------------------------------------------------
__global__ void din_score_kernel(const float* __restrict__ Y2,
                                 const float2* __restrict__ st2,
                                 const float* __restrict__ g2,
                                 const float* __restrict__ be2,
                                 const float* __restrict__ al2,
                                 const float* __restrict__ attW,
                                 const float* __restrict__ attB,
                                 const int* __restrict__ hist,
                                 float* __restrict__ scores) {
  int r = blockIdx.x * 256 + threadIdx.x;
  const float4* yrow = (const float4*)(Y2 + (size_t)r * 16);
  union { float4 q[4]; float f[16]; } y;
  y.q[0] = yrow[0]; y.q[1] = yrow[1]; y.q[2] = yrow[2]; y.q[3] = yrow[3];
  float acc = attB[0];
  #pragma unroll
  for (int k = 0; k < 16; ++k) {
    float x = dice_apply(y.f[k], st2[k], g2[k], be2[k], al2[k]);
    acc += x * attW[k];
  }
  bool m = hist[r] > 0;
  // unmasked: + log(1+eps) (~0 in f32); masked: -inf + log(eps) = -inf
  scores[r] = m ? (acc + logf(1.0f + EPSF)) : -__builtin_inff();
}

// ---------------------------------------------------------------------------
// Per-batch masked softmax + weighted e_hist sum + FC input [e_u | att | e_a].
// ---------------------------------------------------------------------------
__global__ void din_softmax_kernel(const float* __restrict__ scores,
                                   const int* __restrict__ hist,
                                   const int* __restrict__ uids,
                                   const int* __restrict__ tgt,
                                   const float* __restrict__ user_emb,
                                   const float* __restrict__ item_emb,
                                   float* __restrict__ xfc) {   // [B][96]
  __shared__ float ssc[256];
  __shared__ float red[256];
  __shared__ float wgt[256];
  __shared__ float accs[256];
  int b = blockIdx.x, t = threadIdx.x;

  float s = (t < LL) ? scores[(size_t)b * LL + t] : -__builtin_inff();
  ssc[t] = s; red[t] = s;
  __syncthreads();
  for (int off = 128; off > 0; off >>= 1) {
    if (t < off) red[t] = fmaxf(red[t], red[t + off]);
    __syncthreads();
  }
  float mx = red[0];
  __syncthreads();

  float e = (t < LL) ? __expf(ssc[t] - mx) : 0.0f;   // masked rows: exp(-inf)=0
  red[t] = e;
  __syncthreads();
  for (int off = 128; off > 0; off >>= 1) {
    if (t < off) red[t] += red[t + off];
    __syncthreads();
  }
  float S1 = red[0];
  __syncthreads();

  float p = e / S1;
  red[t] = (t < LL) ? p : 0.0f;
  __syncthreads();
  for (int off = 128; off > 0; off >>= 1) {
    if (t < off) red[t] += red[t + off];
    __syncthreads();
  }
  float S2 = red[0];
  __syncthreads();

  wgt[t] = (t < LL) ? (p / (S2 + EPSF)) : 0.0f;
  __syncthreads();

  // weighted sum over history (item_emb = 12.8MB, L2 resident)
  int g = t >> 5, d = t & 31;
  float acc = 0.0f;
  for (int l = g; l < LL; l += 8)
    acc += wgt[l] * item_emb[(size_t)hist[(size_t)b * LL + l] * 32 + d];
  accs[t] = acc;
  __syncthreads();
  if (t < 32) {
    float ea = 0.0f;
    #pragma unroll
    for (int gg = 0; gg < 8; ++gg) ea += accs[gg * 32 + t];
    xfc[(size_t)b * 96 + t]      = user_emb[(size_t)uids[b] * 32 + t];
    xfc[(size_t)b * 96 + 32 + t] = ea;
    xfc[(size_t)b * 96 + 64 + t] = item_emb[(size_t)tgt[b] * 32 + t];
  }
}

// Final: dice on fc1 output, dot with fc_out.
__global__ void din_final_kernel(const float* __restrict__ Yf1,
                                 const float2* __restrict__ st,
                                 const float* __restrict__ g,
                                 const float* __restrict__ be,
                                 const float* __restrict__ al,
                                 const float* __restrict__ Wout,
                                 const float* __restrict__ bout,
                                 float* __restrict__ out) {
  int b = blockIdx.x * blockDim.x + threadIdx.x;
  if (b >= BB) return;
  float acc = bout[0];
  for (int k = 0; k < 80; ++k) {
    float x = dice_apply(Yf1[(size_t)b * 80 + k], st[k], g[k], be[k], al[k]);
    acc += x * Wout[k];
  }
  out[b] = acc;
}

// ---------------------------------------------------------------------------
extern "C" void kernel_launch(void* const* d_in, const int* in_sizes, int n_in,
                              void* d_out, int out_size, void* d_ws, size_t ws_size,
                              hipStream_t stream) {
  (void)in_sizes; (void)n_in; (void)out_size; (void)ws_size;

  const float* user_emb = (const float*)d_in[0];
  const float* item_emb = (const float*)d_in[1];
  const float* a0W = (const float*)d_in[2];  const float* a0b = (const float*)d_in[3];
  const float* a0al = (const float*)d_in[4]; const float* a0g = (const float*)d_in[5];
  const float* a0be = (const float*)d_in[6];
  const float* a1W = (const float*)d_in[7];  const float* a1b = (const float*)d_in[8];
  const float* a1al = (const float*)d_in[9]; const float* a1g = (const float*)d_in[10];
  const float* a1be = (const float*)d_in[11];
  const float* a2W = (const float*)d_in[12]; const float* a2b = (const float*)d_in[13];
  const float* a2al = (const float*)d_in[14];const float* a2g = (const float*)d_in[15];
  const float* a2be = (const float*)d_in[16];
  const float* aoW = (const float*)d_in[17]; const float* aob = (const float*)d_in[18];
  const float* f0W = (const float*)d_in[19]; const float* f0b = (const float*)d_in[20];
  const float* f0al = (const float*)d_in[21];const float* f0g = (const float*)d_in[22];
  const float* f0be = (const float*)d_in[23];
  const float* f1W = (const float*)d_in[24]; const float* f1b = (const float*)d_in[25];
  const float* f1al = (const float*)d_in[26];const float* f1g = (const float*)d_in[27];
  const float* f1be = (const float*)d_in[28];
  const float* foW = (const float*)d_in[29]; const float* fob = (const float*)d_in[30];
  const int* uids = (const int*)d_in[31];
  const int* hist = (const int*)d_in[32];
  const int* hist_lens = (const int*)d_in[33]; (void)hist_lens;
  const int* tgt = (const int*)d_in[34];
  float* out = (float*)d_out;

  // workspace carve-out (256B aligned)
  size_t off = 0;
  auto carve = [&](size_t bytes) -> char* {
    char* p = (char*)d_ws + off;
    off += (bytes + 255) & ~(size_t)255;
    return p;
  };
  unsigned short* Wcomb = (unsigned short*)carve(3 * 4 * 512 * 2);
  unsigned short* W1bf  = (unsigned short*)carve(2 * 2 * 512 * 2);
  unsigned short* W2bf  = (unsigned short*)carve(1 * 512 * 2);
  unsigned short* Wf0bf = (unsigned short*)carve(3 * 13 * 512 * 2);
  unsigned short* Wf1bf = (unsigned short*)carve(7 * 5 * 512 * 2);
  float2* part  = (float2*)carve((size_t)256 * 64 * sizeof(float2));
  float2* st0   = (float2*)carve(64 * sizeof(float2));
  float2* st1   = (float2*)carve(32 * sizeof(float2));
  float2* st2   = (float2*)carve(16 * sizeof(float2));
  float2* stf0  = (float2*)carve(200 * sizeof(float2));
  float2* stf1  = (float2*)carve(80 * sizeof(float2));
  float* xfc    = (float*)carve((size_t)BB * 96 * 4);
  float* Yf0    = (float*)carve((size_t)BB * 200 * 4);
  float* Yf1    = (float*)carve((size_t)BB * 80 * 4);
  float* scores = (float*)carve((size_t)BL * 4);
  float* Y2     = (float*)carve((size_t)BL * 16 * 4);
  float* Y1     = (float*)carve((size_t)BL * 32 * 4);
  float* Y0     = (float*)carve((size_t)BL * 64 * 4);

  auto r32 = [](int c) { return (c + 31) / 32 * 32; };

  // 0) weight prep (fragment-swizzled bf16, layer-0 concat fold)
  din_prepw_kernel<<<1, 256, 0, stream>>>(a0W, a1W, a2W, f0W, f1W,
                                          Wcomb, W1bf, W2bf, Wf0bf, Wf1bf);

  // 1) attention layer 0 (gather + fused concat-GEMM, K=96)  [BL,64]
  din_att0_kernel<<<BL / 16 / 8, 256, 0, stream>>>(item_emb, hist, tgt,
                                                   Wcomb, a0b, Y0);
  din_colstats_part_kernel<<<256, r32(64), 0, stream>>>(Y0, BL, 64, 256, part);
  din_colstats_final_kernel<<<1, r32(64), 0, stream>>>(part, 256, 64, 1.0f / BL, st0);

  // 2) attention layer 1: dice0(Y0) @ W1  [BL,32]
  din_dice_gemm_kernel<2, 64><<<BL / 16 / 8, 256, 8 * 16 * 64 * 2, stream>>>(
      Y0, 64, st0, a0g, a0be, a0al, W1bf, a1b, Y1, 32);
  din_colstats_part_kernel<<<256, r32(32), 0, stream>>>(Y1, BL, 32, 256, part);
  din_colstats_final_kernel<<<1, r32(32), 0, stream>>>(part, 256, 32, 1.0f / BL, st1);

  // 3) attention layer 2: dice1(Y1) @ W2  [BL,16]
  din_dice_gemm_kernel<1, 32><<<BL / 16 / 8, 256, 8 * 16 * 32 * 2, stream>>>(
      Y1, 32, st1, a1g, a1be, a1al, W2bf, a2b, Y2, 16);
  din_colstats_part_kernel<<<256, r32(16), 0, stream>>>(Y2, BL, 16, 256, part);
  din_colstats_final_kernel<<<1, r32(16), 0, stream>>>(part, 256, 16, 1.0f / BL, st2);

  // 4) scores + masked softmax + weighted history sum -> FC input
  din_score_kernel<<<BL / 256, 256, 0, stream>>>(Y2, st2, a2g, a2be, a2al,
                                                 aoW, aob, hist, scores);
  din_softmax_kernel<<<BB, 256, 0, stream>>>(scores, hist, uids, tgt,
                                             user_emb, item_emb, xfc);

  // 5) FC layer 0: xfc @ Wf0  [B,200] (no dice on input)
  din_dice_gemm_kernel<13, 96><<<BB / 16 / 8, 256, 8 * 16 * 96 * 2, stream>>>(
      xfc, 96, nullptr, nullptr, nullptr, nullptr, Wf0bf, f0b, Yf0, 200);
  din_colstats_part_kernel<<<16, r32(200), 0, stream>>>(Yf0, BB, 200, 16, part);
  din_colstats_final_kernel<<<1, r32(200), 0, stream>>>(part, 16, 200, 1.0f / BB, stf0);

  // 6) FC layer 1: dice_f0(Yf0) @ Wf1  [B,80]  (K padded 200 -> 224)
  din_dice_gemm_kernel<5, 224><<<BB / 16 / 8, 256, 8 * 16 * 224 * 2, stream>>>(
      Yf0, 200, stf0, f0g, f0be, f0al, Wf1bf, f1b, Yf1, 80);
  din_colstats_part_kernel<<<16, r32(80), 0, stream>>>(Yf1, BB, 80, 16, part);
  din_colstats_final_kernel<<<1, r32(80), 0, stream>>>(part, 16, 80, 1.0f / BB, stf1);

  // 7) final dice + fc_out dot
  din_final_kernel<<<BB / 256, 256, 0, stream>>>(Yf1, stf1, f1g, f1be, f1al,
                                                 foW, fob, out);
}
